// CausalTransformer_81690277970027
// MI455X (gfx1250) — matrix operand order repaired and checked by reading, not compiled, not run on hardware
//
#include <hip/hip_runtime.h>

// ---------------- model dims ----------------
constexpr int kV = 50257;
constexpr int kC = 1024;
constexpr int kH = 16;
constexpr int kD = 64;          // kC / kH
constexpr int kL = 4;
constexpr int kB = 2;
constexpr int kT = 1024;
constexpr int kM = kB * kT;     // 2048 tokens

typedef __bf16 bf16_t;
typedef __attribute__((ext_vector_type(16))) __bf16 v16bf;
typedef __attribute__((ext_vector_type(8)))  __bf16 v8bf;
typedef __attribute__((ext_vector_type(8)))  float  v8f;

// ---------------- helpers ----------------
__device__ inline v8f zero8() {
  v8f z;
#pragma unroll
  for (int i = 0; i < 8; ++i) z[i] = 0.0f;
  return z;
}

__device__ inline v8f wmma_bf16(v16bf a, v16bf b, v8f c) {
  // D = A(16x32) * B(32x16) + C, f32 accumulate
  return __builtin_amdgcn_wmma_f32_16x16x32_bf16(
      false, a, false, b, (short)0, c, false, false);
}

// A fragment (16x32 bf16, row-major source, ld in elements).
// lane L: row = m0 + (L&15); elems 0..15 = K[k0+off .. +15], off = (L>=16)?16:0
__device__ inline v16bf load_a_frag_bf16(const bf16_t* __restrict__ base, int ld,
                                         int m0, int k0) {
  const int lane = threadIdx.x & 31;
  const bf16_t* p = base + (size_t)(m0 + (lane & 15)) * ld + k0 + ((lane >> 4) ? 16 : 0);
  v8bf lo = *reinterpret_cast<const v8bf*>(p);
  v8bf hi = *reinterpret_cast<const v8bf*>(p + 8);
  return __builtin_shufflevector(lo, hi, 0, 1, 2, 3, 4, 5, 6, 7,
                                 8, 9, 10, 11, 12, 13, 14, 15);
}

// B fragment (32x16) from a row-major [N,K] bf16 matrix (out column n = source row).
// lane L: col = n0 + (L&15); K values = 16 contiguous elems at k0 + ((L>=16)?16:0)
__device__ inline v16bf load_b_frag_bf16(const bf16_t* __restrict__ base, int ld,
                                         int n0, int k0) {
  const int lane = threadIdx.x & 31;
  const bf16_t* p = base + (size_t)(n0 + (lane & 15)) * ld + k0 + ((lane >> 4) ? 16 : 0);
  v8bf lo = *reinterpret_cast<const v8bf*>(p);
  v8bf hi = *reinterpret_cast<const v8bf*>(p + 8);
  return __builtin_shufflevector(lo, hi, 0, 1, 2, 3, 4, 5, 6, 7,
                                 8, 9, 10, 11, 12, 13, 14, 15);
}

// B fragment from a row-major [N,K] f32 weight matrix, converting to bf16 on the fly.
// Row index clamped for the ragged vocab tail.
__device__ inline v16bf load_b_frag_f32(const float* __restrict__ W, int ld, int nmax,
                                        int n0, int k0) {
  const int lane = threadIdx.x & 31;
  int n = n0 + (lane & 15);
  if (n >= nmax) n = nmax - 1;
  const float* p = W + (size_t)n * ld + k0 + ((lane >> 4) ? 16 : 0);
  v16bf b;
#pragma unroll
  for (int q = 0; q < 4; ++q) {
    float4 f = *reinterpret_cast<const float4*>(p + q * 4);
    b[q * 4 + 0] = (bf16_t)f.x;
    b[q * 4 + 1] = (bf16_t)f.y;
    b[q * 4 + 2] = (bf16_t)f.z;
    b[q * 4 + 3] = (bf16_t)f.w;
  }
  return b;
}

__device__ inline float redmax16(float x) {
#pragma unroll
  for (int m = 1; m < 16; m <<= 1) x = fmaxf(x, __shfl_xor(x, m, 32));
  return x;
}
__device__ inline float redsum16(float x) {
#pragma unroll
  for (int m = 1; m < 16; m <<= 1) x += __shfl_xor(x, m, 32);
  return x;
}

// Async global->LDS copy of 32 bytes (two b128s; INST_OFFSET applies to both the LDS
// and the global address). Tracked by ASYNCcnt.
__device__ inline void async_copy32(unsigned lds_off, const void* gsrc) {
  asm volatile(
      "global_load_async_to_lds_b128 %0, %1, off\n\t"
      "global_load_async_to_lds_b128 %0, %1, off offset:16"
      :: "v"(lds_off), "v"(gsrc) : "memory");
}
__device__ inline void wait_async_2() { asm volatile("s_wait_asynccnt 2" ::: "memory"); }
__device__ inline void wait_async_0() { asm volatile("s_wait_asynccnt 0" ::: "memory"); }

// ---------------- kernels ----------------

// x[m,c] = wte[idx[m], c] + wpe[m % T, c]
__global__ void embed_kernel(const int* __restrict__ idx, const float* __restrict__ wte,
                             const float* __restrict__ wpe, float* __restrict__ x) {
  size_t i = (size_t)blockIdx.x * blockDim.x + threadIdx.x;  // over kM*kC
  int c = (int)(i % kC);
  int m = (int)(i / kC);
  int t = m % kT;
  x[i] = wte[(size_t)idx[m] * kC + c] + wpe[(size_t)t * kC + c];
}

// LayerNorm (no bias): out = (x - mu) * rsqrt(var + eps) * w, written bf16.
__global__ void ln_kernel(const float* __restrict__ x, const float* __restrict__ w,
                          bf16_t* __restrict__ out) {
  const int tok = blockIdx.x;
  const int tid = threadIdx.x;
  const float4 xv = *reinterpret_cast<const float4*>(x + (size_t)tok * kC + tid * 4);
  float s  = xv.x + xv.y + xv.z + xv.w;
  float s2 = xv.x * xv.x + xv.y * xv.y + xv.z * xv.z + xv.w * xv.w;
#pragma unroll
  for (int m = 1; m < 32; m <<= 1) {
    s  += __shfl_xor(s,  m, 32);
    s2 += __shfl_xor(s2, m, 32);
  }
  __shared__ float sbuf[16];
  const int wid = tid >> 5;
  if ((tid & 31) == 0) { sbuf[wid] = s; sbuf[8 + wid] = s2; }
  __syncthreads();
  float ts = 0.0f, ts2 = 0.0f;
#pragma unroll
  for (int i = 0; i < 8; ++i) { ts += sbuf[i]; ts2 += sbuf[8 + i]; }
  const float mu = ts * (1.0f / kC);
  const float var = ts2 * (1.0f / kC) - mu * mu;
  const float rs = rsqrtf(var + 1e-5f);
  const float4 wv = *reinterpret_cast<const float4*>(w + tid * 4);
  bf16_t* o = out + (size_t)tok * kC + tid * 4;
  o[0] = (bf16_t)((xv.x - mu) * rs * wv.x);
  o[1] = (bf16_t)((xv.y - mu) * rs * wv.y);
  o[2] = (bf16_t)((xv.z - mu) * rs * wv.z);
  o[3] = (bf16_t)((xv.w - mu) * rs * wv.w);
}

// vT[bh, d, t] = qkv[b, t, 2C + h*64 + d]   (bf16 -> bf16)
__global__ void transv_kernel(const bf16_t* __restrict__ qkv, bf16_t* __restrict__ vT) {
  size_t i = (size_t)blockIdx.x * blockDim.x + threadIdx.x;  // over kB*kH*kD*kT
  int t = (int)(i % kT);
  size_t rest = i / kT;
  int d = (int)(rest % kD);
  int bh = (int)(rest / kD);
  int b = bh / kH, h = bh % kH;
  vT[i] = qkv[((size_t)(b * kT + t)) * (3 * kC) + 2 * kC + h * kD + d];
}

// GEMM: out[m,n] = sum_k A[m,k] * W[n,k]  (+res, +relu); A bf16 [M,K], W f32 [N,K].
// 4-wave workgroup computes a 64x64 block tile. The 64x32 A tile is async-copied to
// LDS (double-buffered, ASYNCcnt) and shared; each wave owns one 16-wide N tile and
// 4 stacked 16x16 accumulators -> 16 wmma per block k-step. The k-loop is peeled so
// the steady state is branch-free (prefetch next stage, wait<=2, compute).
__global__ void gemm_kernel(const bf16_t* __restrict__ A, const float* __restrict__ W,
                            const float* __restrict__ res, float* __restrict__ outF,
                            bf16_t* __restrict__ outB, int N, int K, int relu) {
  const int tid  = threadIdx.x;          // 0..127
  const int wv   = tid >> 5;             // wave id 0..3
  const int lane = tid & 31;
  const int half = lane >> 4;
  const int colL = lane & 15;
  const int mblk = blockIdx.y * 64;
  const int n0 = blockIdx.x * 64 + wv * 16;

  __shared__ __align__(16) bf16_t atile[2][64 * 32];   // [stage][64 rows x 32 k]

  // this thread's 32-byte slice of the A tile
  const int arow = tid >> 1;             // 0..63
  const int akh  = tid & 1;              // which 16-elem k-half
  const unsigned lds_base[2] = {
      (unsigned)(uintptr_t)(void*)&atile[0][arow * 32 + akh * 16],
      (unsigned)(uintptr_t)(void*)&atile[1][arow * 32 + akh * 16]};
  const bf16_t* arow_ptr = A + (size_t)(mblk + arow) * K + akh * 16;

  v8f acc[4];
#pragma unroll
  for (int mt = 0; mt < 4; ++mt) acc[mt] = zero8();

  // prime stage 0
  async_copy32(lds_base[0], arow_ptr);
  int cur = 0;
  // steady state: always a next stage in flight
  for (int k0 = 0; k0 + 32 < K; k0 += 32) {
    async_copy32(lds_base[cur ^ 1], arow_ptr + k0 + 32);
    wait_async_2();            // own async ops for stage `cur` complete
    __syncthreads();           // all waves' slices visible

    v16bf b = load_b_frag_f32(W, K, N, n0, k0);
    v16bf a0 = load_a_frag_bf16(atile[cur], 32, 0, 0);
    v16bf a1 = load_a_frag_bf16(atile[cur], 32, 16, 0);
    v16bf a2 = load_a_frag_bf16(atile[cur], 32, 32, 0);
    v16bf a3 = load_a_frag_bf16(atile[cur], 32, 48, 0);
    acc[0] = wmma_bf16(a0, b, acc[0]);
    acc[1] = wmma_bf16(a1, b, acc[1]);
    acc[2] = wmma_bf16(a2, b, acc[2]);
    acc[3] = wmma_bf16(a3, b, acc[3]);
    __syncthreads();           // done reading stage `cur` before it is overwritten
    cur ^= 1;
  }
  // final stage
  {
    wait_async_0();
    __syncthreads();
    v16bf b = load_b_frag_f32(W, K, N, n0, K - 32);
    v16bf a0 = load_a_frag_bf16(atile[cur], 32, 0, 0);
    v16bf a1 = load_a_frag_bf16(atile[cur], 32, 16, 0);
    v16bf a2 = load_a_frag_bf16(atile[cur], 32, 32, 0);
    v16bf a3 = load_a_frag_bf16(atile[cur], 32, 48, 0);
    acc[0] = wmma_bf16(a0, b, acc[0]);
    acc[1] = wmma_bf16(a1, b, acc[1]);
    acc[2] = wmma_bf16(a2, b, acc[2]);
    acc[3] = wmma_bf16(a3, b, acc[3]);
  }

  const int n = n0 + colL;
  if (n < N) {
#pragma unroll
    for (int mt = 0; mt < 4; ++mt) {
#pragma unroll
      for (int r = 0; r < 8; ++r) {
        const int m = mblk + mt * 16 + r + half * 8;
        float v = acc[mt][r];
        if (res)  v += res[(size_t)m * N + n];
        if (relu) v = fmaxf(v, 0.0f);
        if (outF) outF[(size_t)m * N + n] = v;
        if (outB) outB[(size_t)m * N + n] = (bf16_t)v;
      }
    }
  }
}

// Flash attention, one wave per (b, h, 16-query block). qkv bf16 [M, 3C], vT bf16 [B*H, D, T].
__global__ void attn_kernel(const bf16_t* __restrict__ qkv, const bf16_t* __restrict__ vT,
                            bf16_t* __restrict__ y) {
  const int lane = threadIdx.x & 31;
  const int half = lane >> 4;
  const int colL = lane & 15;
  const int q0 = blockIdx.x * 16;
  const int bh = blockIdx.y;
  const int b = bh / kH, h = bh % kH;
  const int ld = 3 * kC;

  const bf16_t* Qbase = qkv + (size_t)b * kT * ld + h * kD;
  const bf16_t* Kbase = Qbase + kC;
  const bf16_t* Vt = vT + (size_t)bh * kD * kT;   // [D, T]

  const v16bf qf0 = load_a_frag_bf16(Qbase, ld, q0, 0);
  const v16bf qf1 = load_a_frag_bf16(Qbase, ld, q0, 32);

  v8f yacc[4];
#pragma unroll
  for (int j = 0; j < 4; ++j) yacc[j] = zero8();
  float mrow[8], lrow[8];
#pragma unroll
  for (int r = 0; r < 8; ++r) { mrow[r] = -1e30f; lrow[r] = 0.0f; }

  __shared__ __align__(16) bf16_t pbuf[16 * 32];

  for (int kt = 0; kt < q0 + 16; kt += 32) {
    // ---- scores S = Q K^T for two 16-wide key tiles ----
    v8f s0 = zero8(), s1 = zero8();
    {
      v16bf kb0 = load_b_frag_bf16(Kbase, ld, kt, 0);
      v16bf kb1 = load_b_frag_bf16(Kbase, ld, kt, 32);
      v16bf kb2 = load_b_frag_bf16(Kbase, ld, kt + 16, 0);
      v16bf kb3 = load_b_frag_bf16(Kbase, ld, kt + 16, 32);
      s0 = wmma_bf16(qf0, kb0, s0);
      s0 = wmma_bf16(qf1, kb1, s0);
      s1 = wmma_bf16(qf0, kb2, s1);
      s1 = wmma_bf16(qf1, kb3, s1);
    }
    // ---- causal mask + online softmax (rows live across 16 lanes of one half) ----
    float fac[8];
#pragma unroll
    for (int r = 0; r < 8; ++r) {
      const int row = q0 + r + half * 8;
      const int c0 = kt + colL;
      const int c1 = kt + 16 + colL;
      const float e0 = (c0 <= row) ? s0[r] * 0.125f : -1e30f;   // 1/sqrt(64)
      const float e1 = (c1 <= row) ? s1[r] * 0.125f : -1e30f;
      const float bm = redmax16(fmaxf(e0, e1));
      const float mnew = fmaxf(mrow[r], bm);
      fac[r] = __expf(mrow[r] - mnew);
      const float p0 = __expf(e0 - mnew);
      const float p1 = __expf(e1 - mnew);
      lrow[r] = lrow[r] * fac[r] + redsum16(p0 + p1);
      mrow[r] = mnew;
      pbuf[(r + half * 8) * 32 + colL] = (bf16_t)p0;
      pbuf[(r + half * 8) * 32 + 16 + colL] = (bf16_t)p1;
    }
    // rescale running output
#pragma unroll
    for (int j = 0; j < 4; ++j)
#pragma unroll
      for (int r = 0; r < 8; ++r) yacc[j][r] *= fac[r];

    __syncthreads();  // single-wave WG: lowers to waits, orders LDS store->load
    const v16bf pf = load_a_frag_bf16(pbuf, 32, 0, 0);
    __syncthreads();
    // ---- Y += P * V ----
    {
      v16bf bv0 = load_b_frag_bf16(Vt, kT, 0, kt);   // rows = d, contiguous in t
      v16bf bv1 = load_b_frag_bf16(Vt, kT, 16, kt);
      v16bf bv2 = load_b_frag_bf16(Vt, kT, 32, kt);
      v16bf bv3 = load_b_frag_bf16(Vt, kT, 48, kt);
      yacc[0] = wmma_bf16(pf, bv0, yacc[0]);
      yacc[1] = wmma_bf16(pf, bv1, yacc[1]);
      yacc[2] = wmma_bf16(pf, bv2, yacc[2]);
      yacc[3] = wmma_bf16(pf, bv3, yacc[3]);
    }
  }

  // ---- normalize + store bf16 y[b, t, h*64 + d] ----
#pragma unroll
  for (int j = 0; j < 4; ++j)
#pragma unroll
    for (int r = 0; r < 8; ++r) {
      const int row = q0 + r + half * 8;
      const float v = yacc[j][r] / lrow[r];
      y[((size_t)(b * kT + row)) * kC + h * kD + j * 16 + colL] = (bf16_t)v;
    }
}

// ---------------- launcher ----------------
extern "C" void kernel_launch(void* const* d_in, const int* in_sizes, int n_in,
                              void* d_out, int out_size, void* d_ws, size_t ws_size,
                              hipStream_t stream) {
  (void)in_sizes; (void)n_in; (void)out_size; (void)ws_size;
  const int*   idx   = (const int*)d_in[0];
  const float* wte   = (const float*)d_in[1];
  const float* wpe   = (const float*)d_in[2];
  const float* ln1w  = (const float*)d_in[3];
  const float* attnw = (const float*)d_in[4];
  const float* apw   = (const float*)d_in[5];
  const float* ln2w  = (const float*)d_in[6];
  const float* fcw   = (const float*)d_in[7];
  const float* mpw   = (const float*)d_in[8];
  const float* lnfw  = (const float*)d_in[9];
  const float* headw = (const float*)d_in[10];
  float* logits = (float*)d_out;

  char* ws = (char*)d_ws;
  float*  x    = (float*)ws;   ws += (size_t)kM * kC * 4;           // residual stream f32
  bf16_t* lnb  = (bf16_t*)ws;  ws += (size_t)kM * kC * 2;           // LN output bf16
  bf16_t* qkvb = (bf16_t*)ws;  ws += (size_t)kM * 3 * kC * 2;       // qkv bf16
  bf16_t* vT   = (bf16_t*)ws;  ws += (size_t)kB * kH * kD * kT * 2; // V transposed
  bf16_t* yb   = (bf16_t*)ws;  ws += (size_t)kM * kC * 2;           // attn out bf16
  bf16_t* fcb  = (bf16_t*)ws;  ws += (size_t)kM * 4 * kC * 2;       // relu(fc) bf16

  embed_kernel<<<(kM * kC) / 256, 256, 0, stream>>>(idx, wte, wpe, x);

  for (int l = 0; l < kL; ++l) {
    ln_kernel<<<kM, 256, 0, stream>>>(x, ln1w + (size_t)l * kC, lnb);
    gemm_kernel<<<dim3(3 * kC / 64, kM / 64), 128, 0, stream>>>(
        lnb, attnw + (size_t)l * 3 * kC * kC, nullptr, nullptr, qkvb, 3 * kC, kC, 0);
    transv_kernel<<<(kB * kH * kD * kT) / 256, 256, 0, stream>>>(qkvb, vT);
    attn_kernel<<<dim3(kT / 16, kB * kH), 32, 0, stream>>>(qkvb, vT, yb);
    gemm_kernel<<<dim3(kC / 64, kM / 64), 128, 0, stream>>>(
        yb, apw + (size_t)l * kC * kC, x, x, nullptr, kC, kC, 0);
    ln_kernel<<<kM, 256, 0, stream>>>(x, ln2w + (size_t)l * kC, lnb);
    gemm_kernel<<<dim3(4 * kC / 64, kM / 64), 128, 0, stream>>>(
        lnb, fcw + (size_t)l * 4 * kC * kC, nullptr, nullptr, fcb, 4 * kC, kC, 1);
    gemm_kernel<<<dim3(kC / 64, kM / 64), 128, 0, stream>>>(
        fcb, mpw + (size_t)l * kC * 4 * kC, x, x, nullptr, kC, 4 * kC, 0);
  }

  ln_kernel<<<kM, 256, 0, stream>>>(x, lnfw, lnb);
  gemm_kernel<<<dim3((kV + 63) / 64, kM / 64), 128, 0, stream>>>(
      lnb, headw, nullptr, logits, nullptr, kV, kC, 0);
}
